// EncDec_73014444032451
// MI455X (gfx1250) — compile-verified, matrix operand order checked by reading, not run on hardware
//
#include <hip/hip_runtime.h>
#include <hip/hip_bf16.h>

// ---------------------------------------------------------------------------
// Problem constants (from reference): B=512, T=26, F=2048, H=1024, C=22
// ---------------------------------------------------------------------------
#define BB 512
#define TT 26
#define FF 2048
#define HH 1024
#define CC 22
#define GG (4 * HH)   // 4096 gate width

typedef __attribute__((ext_vector_type(16))) __bf16 v16bf;
typedef __attribute__((ext_vector_type(8)))  __bf16 v8bf;
typedef __attribute__((ext_vector_type(8)))  float  v8f;
typedef __attribute__((ext_vector_type(4)))  int    v4i;

// ---- async-to-LDS path availability (gfx1250 ASYNCcnt engine) --------------
#if __has_builtin(__builtin_amdgcn_global_load_async_to_lds_b128) && \
    __has_builtin(__builtin_amdgcn_s_wait_asynccnt)
#define USE_ASYNC_LDS 1
#else
#define USE_ASYNC_LDS 0
#endif

#if USE_ASYNC_LDS
// builtin signature (from hipcc diagnostic): param0 = int4 AS1* (global src),
// param1 = int4 AS3* (LDS dst), then imm offset, imm cpol.
typedef __attribute__((address_space(1))) v4i* gas_ptr;
typedef __attribute__((address_space(3))) v4i* las_ptr;
#define GCAST(p) ((gas_ptr)(p))
#define LCAST(p) ((las_ptr)(p))
#endif

// ---- bf16 <-> f32 helpers (bit-exact, RNE for f32->bf16) -------------------
__device__ __forceinline__ __bf16 f2bf(float f) {
    unsigned u; __builtin_memcpy(&u, &f, 4);
    unsigned r = (u + 0x7FFFu + ((u >> 16) & 1u)) >> 16;
    unsigned short s = (unsigned short)r;
    __bf16 b; __builtin_memcpy(&b, &s, 2);
    return b;
}
__device__ __forceinline__ float bf2f(__bf16 b) {
    unsigned short s; __builtin_memcpy(&s, &b, 2);
    unsigned u = ((unsigned)s) << 16;
    float f; __builtin_memcpy(&f, &u, 4);
    return f;
}

// ---------------------------------------------------------------------------
// Tiled WMMA GEMM:  D[M,N] = A[M,K] * W[N,K]^T  (+ Cin[M,N]) (+ bias[N])
// bf16 operands, f32 accumulation via v_wmma_f32_16x16x32_bf16.
// Block tile 128x64, BK=32, 8 waves (256 thr), each wave a 32x32 sub-tile.
// Double-buffered LDS staged with GLOBAL_LOAD_ASYNC_TO_LDS_B128 (ASYNCcnt),
// one workgroup barrier per K-step.
// Requires: M%128==0, N%64==0, K%32==0 (true for all call sites here).
// ---------------------------------------------------------------------------
#define BM 128
#define BN 64
#define BK 32
#define LDSP 8   // LDS row padding (elements): row stride 80B (16B-aligned)

__global__ __launch_bounds__(256)
void wmma_gemm_bf16(const __bf16* __restrict__ A, int lda,
                    const __bf16* __restrict__ W, int ldw,
                    const float* Cin, long ldcin,      // nullable
                    const float* __restrict__ bias,    // nullable
                    float* D, long ldd,
                    int M, int N, int K)
{
    __shared__ __bf16 As[2][BM][BK + LDSP];
    __shared__ __bf16 Bs[2][BN][BK + LDSP];

    const int tid  = threadIdx.x;
    const int wave = tid >> 5;
    const int lane = tid & 31;
    const int lrow = lane & 15;   // row within 16x16 fragment
    const int kh   = lane >> 4;   // K-half selector for 16-bit A/B layout

    const int bm = blockIdx.y * BM;
    const int bn = blockIdx.x * BN;
    const int wm = (wave & 3) * 32;   // 4 waves along M
    const int wn = (wave >> 2) * 32;  // 2 waves along N

    v8f acc[2][2] = {};

    // staging-load coordinates (per thread)
    const int ar = tid >> 1;            // 0..127 : A row
    const int ah = (tid & 1) * 16;      // 0|16   : A col base (16 elems)
    const int br = tid >> 2;            // 0..63  : W row
    const int bh = (tid & 3) * 8;       // 0..24  : W col base (8 elems)

    const __bf16* __restrict__ aSrc = A + (long)(bm + ar) * lda + ah;
    const __bf16* __restrict__ wSrc = W + (long)(bn + br) * ldw + bh;

#if USE_ASYNC_LDS
    // ---- async (ASYNCcnt) double-buffered staging --------------------------
    // prologue: stage slab 0 into buffer 0
    {
        __builtin_amdgcn_global_load_async_to_lds_b128(
            GCAST(aSrc),     LCAST(&As[0][ar][ah]),     0, 0);
        __builtin_amdgcn_global_load_async_to_lds_b128(
            GCAST(aSrc + 8), LCAST(&As[0][ar][ah + 8]), 0, 0);
        __builtin_amdgcn_global_load_async_to_lds_b128(
            GCAST(wSrc),     LCAST(&Bs[0][br][bh]),     0, 0);
    }
    int buf = 0;
    for (int k0 = 0; k0 < K; k0 += BK) {
        __builtin_amdgcn_s_wait_asynccnt(0);   // my slab-k0 copies landed
        __syncthreads();                       // everyone's copies landed
        if (k0 + BK < K) {                     // stream next slab into other buf
            const __bf16* an  = aSrc + k0 + BK;
            const __bf16* wn2 = wSrc + k0 + BK;
            __builtin_amdgcn_global_load_async_to_lds_b128(
                GCAST(an),     LCAST(&As[buf ^ 1][ar][ah]),     0, 0);
            __builtin_amdgcn_global_load_async_to_lds_b128(
                GCAST(an + 8), LCAST(&As[buf ^ 1][ar][ah + 8]), 0, 0);
            __builtin_amdgcn_global_load_async_to_lds_b128(
                GCAST(wn2),    LCAST(&Bs[buf ^ 1][br][bh]),     0, 0);
        }
#else
    // ---- fallback: sync staging through VGPRs ------------------------------
    const int buf = 0;
    for (int k0 = 0; k0 < K; k0 += BK) {
        if (k0 + BK < K) {  // prefetch next slab -> global_prefetch_b8
            __builtin_prefetch(aSrc + k0 + BK, 0, 1);
            __builtin_prefetch(wSrc + k0 + BK, 0, 1);
        }
        {
            const __bf16* src = aSrc + k0;
            v8bf p0 = *(const v8bf*)(src);
            v8bf p1 = *(const v8bf*)(src + 8);
            *(v8bf*)(&As[0][ar][ah])     = p0;
            *(v8bf*)(&As[0][ar][ah + 8]) = p1;
        }
        { *(v8bf*)(&Bs[0][br][bh]) = *(const v8bf*)(wSrc + k0); }
        __syncthreads();
#endif

        // ---- build fragments per documented 16-bit layout ------------------
        // lane holds row (l%16); VGPR0-3 <- K in [kh*8, kh*8+8),
        //                        VGPR4-7 <- K in [16+kh*8, 16+kh*8+8)
        v16bf afrag[2], bfrag[2];
#pragma unroll
        for (int i = 0; i < 2; ++i) {
            const __bf16* ap = &As[buf][wm + i * 16 + lrow][kh * 8];
            v8bf lo = *(const v8bf*)(ap);
            v8bf hi = *(const v8bf*)(ap + 16);
            afrag[i] = __builtin_shufflevector(lo, hi,
                0,1,2,3,4,5,6,7,8,9,10,11,12,13,14,15);
        }
#pragma unroll
        for (int j = 0; j < 2; ++j) {
            const __bf16* bp = &Bs[buf][wn + j * 16 + lrow][kh * 8];
            v8bf lo = *(const v8bf*)(bp);
            v8bf hi = *(const v8bf*)(bp + 16);
            bfrag[j] = __builtin_shufflevector(lo, hi,
                0,1,2,3,4,5,6,7,8,9,10,11,12,13,14,15);
        }

#pragma unroll
        for (int i = 0; i < 2; ++i)
#pragma unroll
            for (int j = 0; j < 2; ++j)
                acc[i][j] = __builtin_amdgcn_wmma_f32_16x16x32_bf16(
                    false, afrag[i], false, bfrag[j],
                    (short)0, acc[i][j], false, false);

#if USE_ASYNC_LDS
        buf ^= 1;   // double-buffered: no trailing barrier needed
#else
        __syncthreads();
#endif
    }

    // ---- epilogue: C/D layout -> lane holds col l%16, rows v + 8*(l>>4) ----
#pragma unroll
    for (int i = 0; i < 2; ++i) {
#pragma unroll
        for (int j = 0; j < 2; ++j) {
            const int col   = bn + wn + j * 16 + lrow;
            const int mrow0 = bm + wm + i * 16 + kh * 8;
            const float badd = bias ? bias[col] : 0.0f;
#pragma unroll
            for (int v = 0; v < 8; ++v) {
                float val = acc[i][j][v] + badd;
                const long m = (long)(mrow0 + v);
                if (Cin) val += Cin[m * ldcin + col];
                D[m * ldd + col] = val;
            }
        }
    }
}

// ---------------------------------------------------------------------------
// Elementwise / small kernels
// ---------------------------------------------------------------------------
__global__ void cvt_f32_bf16(const float* __restrict__ s, __bf16* __restrict__ d, long n) {
    long i = (long)blockIdx.x * blockDim.x + threadIdx.x;
    long stride = (long)gridDim.x * blockDim.x;
    for (; i < n; i += stride) d[i] = f2bf(s[i]);
}
__global__ void zero_f32(float* p, long n) {
    long i = (long)blockIdx.x * blockDim.x + threadIdx.x;
    long stride = (long)gridDim.x * blockDim.x;
    for (; i < n; i += stride) p[i] = 0.0f;
}
__global__ void zero_bf16(__bf16* p, long n) {
    long i = (long)blockIdx.x * blockDim.x + threadIdx.x;
    long stride = (long)gridDim.x * blockDim.x;
    for (; i < n; i += stride) p[i] = f2bf(0.0f);
}
__global__ void vec_add(const float* __restrict__ a, const float* __restrict__ b,
                        float* __restrict__ o, int n) {
    int i = blockIdx.x * blockDim.x + threadIdx.x;
    if (i < n) o[i] = a[i] + b[i];
}

// gates [B,4H] in i,f,g,o order; c f32 state; h bf16 out; optional h_ts slice
__global__ void lstm_cell(const float* __restrict__ gates, float* __restrict__ c,
                          __bf16* __restrict__ h, __bf16* hts, long hts_stride) {
    int i = blockIdx.x * blockDim.x + threadIdx.x;
    if (i >= BB * HH) return;
    const int b = i >> 10;        // /H
    const int j = i & (HH - 1);   // %H
    const long gb = (long)b * GG;
    const float ig = gates[gb + j];
    const float fg = gates[gb + HH + j];
    const float gg = gates[gb + 2 * HH + j];
    const float og = gates[gb + 3 * HH + j];
    const float si = 1.0f / (1.0f + expf(-ig));
    const float sf = 1.0f / (1.0f + expf(-fg));
    const float so = 1.0f / (1.0f + expf(-og));
    const float cn = sf * c[i] + si * tanhf(gg);
    c[i] = cn;
    const float hv = so * tanhf(cn);
    const __bf16 hb = f2bf(hv);
    h[i] = hb;
    if (hts) hts[(long)b * hts_stride + j] = hb;
}

// attention: one 256-thread block per batch element; dec_h [B,H] bf16,
// ph [B,T,H] f32 -> attn [B,H] bf16
__global__ __launch_bounds__(256)
void attention_kernel(const __bf16* __restrict__ h, const float* __restrict__ ph,
                      __bf16* __restrict__ attn, float inv_sqrt_h) {
    __shared__ float red[256];
    __shared__ float wts[TT];
    const int b   = blockIdx.x;
    const int tid = threadIdx.x;
    float hv[4];
#pragma unroll
    for (int e = 0; e < 4; ++e) hv[e] = bf2f(h[(long)b * HH + tid * 4 + e]);
    const float* phb = ph + (long)b * TT * HH;

    for (int t = 0; t < TT; ++t) {
        const float* row = phb + (long)t * HH + tid * 4;
        float p = hv[0] * row[0] + hv[1] * row[1] + hv[2] * row[2] + hv[3] * row[3];
        red[tid] = p;
        __syncthreads();
        for (int s = 128; s > 0; s >>= 1) {
            if (tid < s) red[tid] += red[tid + s];
            __syncthreads();
        }
        if (tid == 0) wts[t] = red[0] * inv_sqrt_h;
        __syncthreads();
    }
    if (tid == 0) {
        float mx = -1e30f;
        for (int t = 0; t < TT; ++t) mx = fmaxf(mx, wts[t]);
        float s = 0.0f;
        for (int t = 0; t < TT; ++t) { float e = expf(wts[t] - mx); wts[t] = e; s += e; }
        const float inv = 1.0f / s;
        for (int t = 0; t < TT; ++t) wts[t] *= inv;
    }
    __syncthreads();
#pragma unroll
    for (int e = 0; e < 4; ++e) {
        const int idx = tid * 4 + e;
        float a = 0.0f;
        for (int t = 0; t < TT; ++t) a += wts[t] * phb[(long)t * HH + idx];
        attn[(long)b * HH + idx] = f2bf(a);
    }
}

// classifier: one thread per (b,c); out pre-offset to timestep slice
__global__ void classifier_kernel(const __bf16* __restrict__ h,
                                  const float* __restrict__ Wc,
                                  const float* __restrict__ bc,
                                  float* __restrict__ out) {
    int i = blockIdx.x * blockDim.x + threadIdx.x;
    if (i >= BB * CC) return;
    const int b = i / CC, c = i % CC;
    const __bf16* hb = h + (long)b * HH;
    const float*  wr = Wc + (long)c * HH;
    float s = bc[c];
    for (int k = 0; k < HH; ++k) s += bf2f(hb[k]) * wr[k];
    out[(long)b * TT * CC + c] = s;
}

// ---------------------------------------------------------------------------
// Host orchestration
// ---------------------------------------------------------------------------
extern "C" void kernel_launch(void* const* d_in, const int* in_sizes, int n_in,
                              void* d_out, int out_size, void* d_ws, size_t ws_size,
                              hipStream_t stream) {
    (void)in_sizes; (void)n_in; (void)out_size; (void)ws_size;
    const float* x     = (const float*)d_in[0];
    const float* eWih  = (const float*)d_in[1];
    const float* eWhh  = (const float*)d_in[2];
    const float* ebih  = (const float*)d_in[3];
    const float* ebhh  = (const float*)d_in[4];
    const float* projW = (const float*)d_in[5];
    const float* projb = (const float*)d_in[6];
    const float* dWih  = (const float*)d_in[7];
    const float* dWhh  = (const float*)d_in[8];
    const float* dbih  = (const float*)d_in[9];
    const float* dbhh  = (const float*)d_in[10];
    const float* clsW  = (const float*)d_in[11];
    const float* clsb  = (const float*)d_in[12];
    float* out = (float*)d_out;

    // ---- workspace carve-up ----
    char* ws = (char*)d_ws;
    size_t off = 0;
    auto carve = [&](size_t bytes) -> char* {
        char* p = ws + off;
        off += (bytes + 255) & ~(size_t)255;
        return p;
    };
    __bf16* x_bf     = (__bf16*)carve((size_t)BB * TT * FF * 2);
    __bf16* eWih_bf  = (__bf16*)carve((size_t)GG * FF * 2);
    __bf16* eWhh_bf  = (__bf16*)carve((size_t)GG * HH * 2);
    __bf16* projW_bf = (__bf16*)carve((size_t)HH * HH * 2);
    __bf16* dWih_bf  = (__bf16*)carve((size_t)GG * (FF + HH) * 2);
    __bf16* dWhh_bf  = (__bf16*)carve((size_t)GG * HH * 2);
    float*  xg       = (float*)carve((size_t)BB * TT * GG * 4);
    float*  dxg      = (float*)carve((size_t)BB * TT * GG * 4);
    __bf16* h_bf     = (__bf16*)carve((size_t)BB * HH * 2);
    float*  c_st     = (float*)carve((size_t)BB * HH * 4);
    __bf16* hts_bf   = (__bf16*)carve((size_t)BB * TT * HH * 2);
    float*  ph       = (float*)carve((size_t)BB * TT * HH * 4);
    float*  gates    = (float*)carve((size_t)BB * GG * 4);
    __bf16* attn_bf  = (__bf16*)carve((size_t)BB * HH * 2);
    float*  ebsum    = (float*)carve((size_t)GG * 4);
    float*  dbsum    = (float*)carve((size_t)GG * 4);

    const int CVTB = 4096;
    // ---- precision conversion (f32 -> bf16) ----
    cvt_f32_bf16<<<CVTB, 256, 0, stream>>>(x,     x_bf,     (long)BB * TT * FF);
    cvt_f32_bf16<<<CVTB, 256, 0, stream>>>(eWih,  eWih_bf,  (long)GG * FF);
    cvt_f32_bf16<<<CVTB, 256, 0, stream>>>(eWhh,  eWhh_bf,  (long)GG * HH);
    cvt_f32_bf16<<<CVTB, 256, 0, stream>>>(projW, projW_bf, (long)HH * HH);
    cvt_f32_bf16<<<CVTB, 256, 0, stream>>>(dWih,  dWih_bf,  (long)GG * (FF + HH));
    cvt_f32_bf16<<<CVTB, 256, 0, stream>>>(dWhh,  dWhh_bf,  (long)GG * HH);
    vec_add<<<(GG + 255) / 256, 256, 0, stream>>>(ebih, ebhh, ebsum, GG);
    vec_add<<<(GG + 255) / 256, 256, 0, stream>>>(dbih, dbhh, dbsum, GG);

    auto gemm = [&](const __bf16* A, int lda, const __bf16* Wm, int ldw,
                    const float* Cin, long ldcin, const float* bias,
                    float* Dm, long ldd, int M, int N, int K) {
        dim3 g(N / BN, M / BM);
        wmma_gemm_bf16<<<g, 256, 0, stream>>>(A, lda, Wm, ldw, Cin, ldcin,
                                              bias, Dm, ldd, M, N, K);
    };

    // ---- hoisted input-path GEMMs (parallel over all B*T rows) ----
    // xg  = x @ enc_Wih^T + (enc_bih + enc_bhh)                [B*T, 4H]
    gemm(x_bf, FF, eWih_bf, FF, nullptr, 0, ebsum, xg, GG, BB * TT, GG, FF);
    // dxg = x @ dec_Wih[:, :F]^T + (dec_bih + dec_bhh)         [B*T, 4H]
    gemm(x_bf, FF, dWih_bf, FF + HH, nullptr, 0, dbsum, dxg, GG, BB * TT, GG, FF);

    // ---- encoder recurrence ----
    zero_bf16<<<CVTB, 256, 0, stream>>>(h_bf, (long)BB * HH);
    zero_f32 <<<CVTB, 256, 0, stream>>>(c_st, (long)BB * HH);
    for (int t = 0; t < TT; ++t) {
        // gates = h @ enc_Whh^T + xg[:, t, :]
        gemm(h_bf, HH, eWhh_bf, HH, xg + (long)t * GG, (long)TT * GG, nullptr,
             gates, GG, BB, GG, HH);
        lstm_cell<<<(BB * HH) / 256, 256, 0, stream>>>(
            gates, c_st, h_bf, hts_bf + (long)t * HH, (long)TT * HH);
    }

    // ---- projection of all encoder states ----
    // ph = h_ts @ proj_W^T + proj_b                            [B*T, H]
    gemm(hts_bf, HH, projW_bf, HH, nullptr, 0, projb, ph, HH, BB * TT, HH, HH);

    // ---- decoder recurrence with attention ----
    zero_bf16<<<CVTB, 256, 0, stream>>>(h_bf, (long)BB * HH);
    zero_f32 <<<CVTB, 256, 0, stream>>>(c_st, (long)BB * HH);
    const float inv_sqrt_h = 0.03125f;  // 1/sqrt(1024)
    for (int t = 0; t < TT; ++t) {
        attention_kernel<<<BB, 256, 0, stream>>>(h_bf, ph, attn_bf, inv_sqrt_h);
        // gates  = attn @ dec_Wih[:, F:]^T + dxg[:, t, :]
        gemm(attn_bf, HH, dWih_bf + FF, FF + HH, dxg + (long)t * GG,
             (long)TT * GG, nullptr, gates, GG, BB, GG, HH);
        // gates += h @ dec_Whh^T   (in-place accumulate)
        gemm(h_bf, HH, dWhh_bf, HH, gates, GG, nullptr, gates, GG, BB, GG, HH);
        lstm_cell<<<(BB * HH) / 256, 256, 0, stream>>>(
            gates, c_st, h_bf, (__bf16*)nullptr, 0);
        classifier_kernel<<<(BB * CC + 255) / 256, 256, 0, stream>>>(
            h_bf, clsW, clsb, out + (long)t * CC);
    }
}
